// Encoder_meta_69157563401011
// MI455X (gfx1250) — matrix operand order repaired and checked by reading, not compiled, not run on hardware
//
#include <hip/hip_runtime.h>

typedef __attribute__((ext_vector_type(16))) __bf16 v16bf;
typedef __attribute__((ext_vector_type(8)))  float  v8f;

#define B_  256
#define T_  336
#define KE_ 16
#define H_  512
#define P_  256

#define NBLK 64
#define NTHR 256
#define NWAVE ((NBLK * NTHR) / 32)   // 512 waves

// ---- workspace layout (bytes) ----
#define OFF_WHT    0u         // WhT bf16 [P_,H_]            262144
#define OFF_WIH    262144u    // W_ih bf16 [3H_,P_]          786432
#define OFF_WHH    1048576u   // W_hh bf16 [3H_,H_]         1572864
#define OFF_HF0    2621440u   // h fp32 buf0/buf1            524288 each
#define OFF_HBF0   3670016u   // h bf16 buf0/buf1            262144 each
#define OFF_XPRIME 4194304u   // x' bf16 [B_,P_]             131072
#define OFF_GHS    4325376u   // gh fp32 [B_,3H_]           1572864
#define OFF_XIN    5898240u   // xin bf16 [B_,32] (K-pad)     16384
#define OFF_WXT    5914624u   // WxT bf16 [P_,32] (K-pad)     16384
#define OFF_SYNC   5931008u   // grid-sync counter                4

#define HF_BYTES   524288u
#define HBF_BYTES  262144u

__device__ __forceinline__ unsigned short f2bf(float f) {
  unsigned int u = __float_as_uint(f);
  u += 0x7FFFu + ((u >> 16) & 1u);   // round-to-nearest-even
  return (unsigned short)(u >> 16);
}

__device__ __forceinline__ float sigmoidf(float x) {
  return 1.0f / (1.0f + __expf(-x));
}

// A-fragment (16x32 bf16, MxK) k = 16*(lane/16) + s, same order as load_b_frag.
__device__ __forceinline__ v16bf load_a_frag(const unsigned short* A, int row0,
                                             int k0, int lda, int lane) {
  const int m = lane & 15, g = lane >> 4;
  const unsigned int* p = (const unsigned int*)(A + (size_t)(row0 + m) * lda + k0);
  union { unsigned int u[8]; v16bf v; } r;
#pragma unroll
  for (int i = 0; i < 4; ++i) {
    r.u[i]     = p[i + 8 * g];
    r.u[i + 4] = p[i + 4 + 8 * g];
  }
  return r.v;
}

// B-fragment (32x16 bf16, KxN) from K-contiguous buffer Bt[n*ldb + k].
__device__ __forceinline__ v16bf load_b_frag(const unsigned short* Bt, int n0,
                                             int k0, int ldb, int lane) {
  const int m = lane & 15, g = lane >> 4;
  const unsigned int* p = (const unsigned int*)(Bt + (size_t)(n0 + m) * ldb + k0 + 16 * g);
  union { unsigned int u[8]; v16bf v; } r;
#pragma unroll
  for (int i = 0; i < 8; ++i) r.u[i] = p[i];
  return r.v;
}

__device__ __forceinline__ v8f bf16_wmma(v16bf a, v16bf b, v8f c) {
  return __builtin_amdgcn_wmma_f32_16x16x32_bf16(false, a, false, b, (short)0, c,
                                                 false, false);
}

// Grid-wide barrier: monotonically increasing target, no counter reset.
__device__ __forceinline__ void gridsync(unsigned int* cnt, unsigned int* target) {
  __syncthreads();
  if (threadIdx.x == 0) {
    __threadfence();  // make this block's global writes visible device-wide
    __hip_atomic_fetch_add(cnt, 1u, __ATOMIC_ACQ_REL, __HIP_MEMORY_SCOPE_AGENT);
    while (__hip_atomic_load(cnt, __ATOMIC_ACQUIRE, __HIP_MEMORY_SCOPE_AGENT) < *target)
      __builtin_amdgcn_s_sleep(1);
  }
  __syncthreads();
  *target += NBLK;
}

__global__ void k_zero_sync(unsigned int* c) { *c = 0u; }

// Persistent kernel: phase 0 = weight prep; per step:
//   phase 1: gh = h@W_hh^T + b_hh (1536 tiles)  and  x' = tanh(h@Wh + xin@Wx + b_t) (256 tiles)
//   phase 2: gx = x'@W_ih^T, gate math, h update (512 tiles) + build xin(t+1)
__global__ __launch_bounds__(NTHR) void k_gru_persistent(
    const float* __restrict__ xl, const float* __restrict__ xe,
    const float* __restrict__ Wh, const float* __restrict__ Wx,
    const float* __restrict__ bt, const float* __restrict__ Wih,
    const float* __restrict__ Whh, const float* __restrict__ bih,
    const float* __restrict__ bhh, float* __restrict__ out, char* ws) {
  // All workspace pointers are straight-line offsets from the kernel arg so
  // InferAddressSpaces keeps every access on the GLOBAL path (no flat_load).
  unsigned short* WhT    = (unsigned short*)(ws + OFF_WHT);
  unsigned short* Wihb   = (unsigned short*)(ws + OFF_WIH);
  unsigned short* Whhb   = (unsigned short*)(ws + OFF_WHH);
  float*          hf0    = (float*)(ws + OFF_HF0);
  unsigned short* hbf0   = (unsigned short*)(ws + OFF_HBF0);
  unsigned short* xprime = (unsigned short*)(ws + OFF_XPRIME);
  float*          ghs    = (float*)(ws + OFF_GHS);
  unsigned short* xinbf  = (unsigned short*)(ws + OFF_XIN);
  unsigned short* WxT    = (unsigned short*)(ws + OFF_WXT);
  unsigned int*   scnt   = (unsigned int*)(ws + OFF_SYNC);

  const int tid  = blockIdx.x * NTHR + threadIdx.x;
  const int lane = threadIdx.x & 31;
  const int wave = tid >> 5;
  const int m = lane & 15, g = lane >> 4;

  // ---------------- phase 0: weight conversion + state init ----------------
  for (int i = tid; i < P_ * H_; i += NBLK * NTHR) {        // WhT[n*H+k] = Wh[k][n]
    int n = i >> 9, k = i & (H_ - 1);
    WhT[i] = f2bf(Wh[k * P_ + n]);
  }
  for (int i = tid; i < 3 * H_ * P_; i += NBLK * NTHR) Wihb[i] = f2bf(Wih[i]);
  for (int i = tid; i < 3 * H_ * H_; i += NBLK * NTHR) Whhb[i] = f2bf(Whh[i]);
  for (int i = tid; i < P_ * 32; i += NBLK * NTHR) {        // WxT[n*32+k], K padded to 32
    int n = i >> 5, k = i & 31;
    WxT[i] = (k <= KE_) ? f2bf(Wx[k * P_ + n]) : (unsigned short)0;
  }
  for (int i = tid; i < B_ * H_; i += NBLK * NTHR) { hf0[i] = 0.0f; hbf0[i] = 0; }
  for (int i = tid; i < B_ * 32; i += NBLK * NTHR) {        // xin tile for t = 0
    int r = i >> 5, k = i & 31;
    float v = (k == 0) ? xl[r * T_] : (k <= KE_ ? xe[(size_t)r * T_ * KE_ + (k - 1)] : 0.0f);
    xinbf[i] = f2bf(v);
  }
  unsigned int target = NBLK;
  gridsync(scnt, &target);

  // ---------------- sequential scan over T ----------------
  for (int t = 0; t < T_; ++t) {
    // ping-pong via integer offset math (no pointer arrays -> no alloca -> no flat)
    const unsigned cb = (unsigned)(t & 1);
    const unsigned short* hbf_c = (const unsigned short*)(ws + OFF_HBF0 + cb * HBF_BYTES);
    const float*          hf_c  = (const float*)(ws + OFF_HF0 + cb * HF_BYTES);
    unsigned short*       hbf_n = (unsigned short*)(ws + OFF_HBF0 + (cb ^ 1u) * HBF_BYTES);
    float*                hf_n  = (float*)(ws + OFF_HF0 + (cb ^ 1u) * HF_BYTES);

    // -------- phase 1: 1536 gh tiles + 256 x' tiles, round-robin over 512 waves --------
    for (int item = wave; item < 1792; item += NWAVE) {
      if (item < 1536) {
        const int tM = item / 96, tNc = item % 96;          // gh: [256, 1536]
        const int row0 = tM * 16, col0 = tNc * 16;
        v8f c = {};
#pragma unroll
        for (int k0 = 0; k0 < H_; k0 += 32)
          c = bf16_wmma(load_a_frag(hbf_c, row0, k0, H_, lane),
                        load_b_frag(Whhb, col0, k0, H_, lane), c);
        const int col = col0 + m;
#pragma unroll
        for (int v = 0; v < 8; ++v)
          ghs[(size_t)(row0 + v + 8 * g) * (3 * H_) + col] = c[v] + bhh[col];
      } else {
        const int q = item - 1536;                          // x': [256, 256]
        const int row0 = (q >> 4) * 16, n0 = (q & 15) * 16;
        v8f c = {};
#pragma unroll
        for (int k0 = 0; k0 < H_; k0 += 32)
          c = bf16_wmma(load_a_frag(hbf_c, row0, k0, H_, lane),
                        load_b_frag(WhT, n0, k0, H_, lane), c);
        // K-padded xin @ Wx as one extra WMMA (K = 32, rows 17..31 zero)
        c = bf16_wmma(load_a_frag(xinbf, row0, 0, 32, lane),
                      load_b_frag(WxT, n0, 0, 32, lane), c);
        const int col = n0 + m;
#pragma unroll
        for (int v = 0; v < 8; ++v) {
          const int row = row0 + v + 8 * g;
          xprime[row * P_ + col] = f2bf(tanhf(c[v] + bt[col]));
        }
      }
    }
    gridsync(scnt, &target);

    // -------- phase 2: 512 h_new tiles (one per wave) --------
    {
      const int tM = wave >> 5, tN = wave & 31;
      const int row0 = tM * 16, n0 = tN * 16;
      v8f cr = {}, cz = {}, cn = {};
#pragma unroll
      for (int k0 = 0; k0 < P_; k0 += 32) {
        v16bf a = load_a_frag(xprime, row0, k0, P_, lane);
        cr = bf16_wmma(a, load_b_frag(Wihb,             n0, k0, P_, lane), cr);
        cz = bf16_wmma(a, load_b_frag(Wihb + 1*H_*P_,   n0, k0, P_, lane), cz);
        cn = bf16_wmma(a, load_b_frag(Wihb + 2*H_*P_,   n0, k0, P_, lane), cn);
      }
      const int col = n0 + m;
      float* dstf = (t == T_ - 1) ? out : hf_n;  // select of two global ptrs: infers fine
#pragma unroll
      for (int v = 0; v < 8; ++v) {
        const int row = row0 + v + 8 * g;
        const float* ghr = ghs + (size_t)row * (3 * H_) + col;
        float r  = sigmoidf(cr[v] + bih[col]          + ghr[0]);
        float z  = sigmoidf(cz[v] + bih[col + H_]     + ghr[H_]);
        float nn = tanhf(cn[v] + bih[col + 2 * H_] + r * ghr[2 * H_]);
        float hp = hf_c[(size_t)row * H_ + col];
        float hv = (1.0f - z) * nn + z * hp;
        dstf[(size_t)row * H_ + col]  = hv;
        hbf_n[(size_t)row * H_ + col] = f2bf(hv);
      }
      if (t + 1 < T_) {  // build xin tile for step t+1 (state-independent)
        for (int i = tid; i < B_ * 32; i += NBLK * NTHR) {
          int r = i >> 5, k = i & 31;
          float v = (k == 0) ? xl[r * T_ + t + 1]
                             : (k <= KE_ ? xe[((size_t)r * T_ + t + 1) * KE_ + (k - 1)] : 0.0f);
          xinbf[i] = f2bf(v);
        }
      }
    }
    gridsync(scnt, &target);
  }
}

// ---------------- host launcher ----------------
extern "C" void kernel_launch(void* const* d_in, const int* in_sizes, int n_in,
                              void* d_out, int out_size, void* d_ws, size_t ws_size,
                              hipStream_t stream) {
  const float* xl  = (const float*)d_in[0];
  const float* xe  = (const float*)d_in[1];
  const float* Wh  = (const float*)d_in[2];
  const float* Wx  = (const float*)d_in[3];
  const float* bt  = (const float*)d_in[4];
  const float* Wih = (const float*)d_in[5];
  const float* Whh = (const float*)d_in[6];
  const float* bih = (const float*)d_in[7];
  const float* bhh = (const float*)d_in[8];
  char* ws = (char*)d_ws;

  k_zero_sync<<<1, 1, 0, stream>>>((unsigned int*)(ws + OFF_SYNC));
  // 64 blocks x 256 threads: fully resident on MI455X, 512 concurrent WMMA waves.
  k_gru_persistent<<<NBLK, NTHR, 0, stream>>>(xl, xe, Wh, Wx, bt, Wih, Whh,
                                              bih, bhh, (float*)d_out, ws);
}
